// HeteroEdgeAwareTransformer_17265768530360
// MI455X (gfx1250) — matrix operand order repaired and checked by reading, not compiled, not run on hardware
//
#include <hip/hip_runtime.h>
#include <stddef.h>

// ---------------------------------------------------------------------------
// HeteroEdgeAwareTransformer for MI455X (gfx1250, wave32)
// Dense node GEMMs via v_wmma_f32_16x16x32_bf16 with fragment-packed weights;
// each wave computes a 32x64 tile (2 M-subtiles share B frags -> 8 wmma/k-step).
// Edge softmax via atomics with float4 gathers.
// ---------------------------------------------------------------------------

typedef __attribute__((ext_vector_type(16))) __bf16          v16bf;
typedef __attribute__((ext_vector_type(16))) unsigned short  v16us;
typedef __attribute__((ext_vector_type(8)))  float           v8f;
typedef __attribute__((ext_vector_type(4)))  unsigned int    v4u;
typedef __attribute__((ext_vector_type(4)))  float           v4f;

union Frag16 { v16us u; v16bf h; v4u q[2]; };

__device__ __forceinline__ unsigned short f2bf(float f) {
    unsigned int x = __float_as_uint(f);
    unsigned int r = x + 0x7fffu + ((x >> 16) & 1u);   // round-to-nearest-even
    return (unsigned short)(r >> 16);
}

__device__ __forceinline__ void atomicMaxF(float* addr, float val) {
    int old = __float_as_int(*addr);
    while (__int_as_float(old) < val) {
        int assumed = old;
        old = atomicCAS((int*)addr, assumed, __float_as_int(val));
        if (old == assumed) break;
    }
}

// ---------------------------------------------------------------------------
// Weight convert + pack into WMMA B-fragment order.
// B[K,N] f32 -> packed bf16 fragments: frag (kt,nt) covers K rows kt*32..+31,
// cols nt*16..+15. Per fragment: 32 lanes x 16 bf16 contiguous (32B/lane).
// element j of lane L: K = kt*32 + (L>>4)*16 + j, col = nt*16 + (L&15).
// ---------------------------------------------------------------------------
__global__ void pack_b_wmma(const float* __restrict__ W, unsigned short* __restrict__ P,
                            int K, int N)
{
    int t = blockIdx.x * 256 + threadIdx.x;
    if (t >= K * N) return;
    int j       = t & 15;
    int lane    = (t >> 4) & 31;
    int frag    = t >> 9;                 // kt*(N/16) + nt
    int tiles16 = N >> 4;
    int kt = frag / tiles16, nt = frag - kt * tiles16;
    int kk = kt * 32 + (lane >> 4) * 16 + j;
    int cc = nt * 16 + (lane & 15);
    P[t] = f2bf(W[(size_t)kk * N + cc]);
}

// ---------------------------------------------------------------------------
// WMMA GEMM: C[M,N](f32) = A[M,K](bf16 row-major) @ Bp(packed frags) + bias
// N in {64,256}, K multiple of 32. 4 waves/block, one 32x64 tile per wave.
// ---------------------------------------------------------------------------
__global__ __launch_bounds__(128)
void gemm_bf16_wmma(const unsigned short* __restrict__ A,
                    const unsigned short* __restrict__ Bp,
                    const float* __restrict__ bias,
                    float* __restrict__ C,
                    int M, int N, int K, int accumulate)
{
    const int lane   = threadIdx.x & 31;
    const int wave   = threadIdx.x >> 5;
    const int tilesN = N >> 6;
    const int tile   = blockIdx.x * 4 + wave;
    const int tm     = tile / tilesN;      // 32-row tile index
    const int tn     = tile - tm * tilesN;
    if (tm * 32 >= M) return;

    const int half = lane >> 4;            // 0: lanes 0-15, 1: lanes 16-31
    const int r    = lane & 15;
    int arow0 = tm * 32 + r;      if (arow0 >= M) arow0 = M - 1;
    int arow1 = tm * 32 + 16 + r; if (arow1 >= M) arow1 = M - 1;
    const unsigned short* A0 = A + (size_t)arow0 * K;
    const unsigned short* A1 = A + (size_t)arow1 * K;

    const int tiles16 = N >> 4;
    v8f a0 = {}, a1 = {}, a2 = {}, a3 = {};   // rows tm*32 .. +15
    v8f c0 = {}, c1 = {}, c2 = {}, c3 = {};   // rows tm*32+16 .. +31

    for (int ks = 0, kt = 0; ks < K; ks += 32, ++kt) {
        // A fragments (16x32 bf16 each): contiguous 16 k per lane half (k = 16*half + s), 16B aligned
        Frag16 afA, afB;
        const unsigned short* ap0 = A0 + ks + half * 16;
        const unsigned short* ap1 = A1 + ks + half * 16;
        afA.q[0] = *(const v4u*)(ap0);
        afA.q[1] = *(const v4u*)(ap0 + 8);
        afB.q[0] = *(const v4u*)(ap1);
        afB.q[1] = *(const v4u*)(ap1 + 8);

        // B fragments from packed buffer: 512 ushorts per fragment
        const unsigned short* bp =
            Bp + ((size_t)(kt * tiles16 + tn * 4) * 32 + lane) * 16;
        Frag16 b0, b1, b2, b3;
        b0.q[0] = ((const v4u*)bp)[0];          b0.q[1] = ((const v4u*)bp)[1];
        b1.q[0] = ((const v4u*)(bp + 512))[0];  b1.q[1] = ((const v4u*)(bp + 512))[1];
        b2.q[0] = ((const v4u*)(bp + 1024))[0]; b2.q[1] = ((const v4u*)(bp + 1024))[1];
        b3.q[0] = ((const v4u*)(bp + 1536))[0]; b3.q[1] = ((const v4u*)(bp + 1536))[1];

        a0 = __builtin_amdgcn_wmma_f32_16x16x32_bf16(false, afA.h, false, b0.h, (short)0, a0, false, false);
        a1 = __builtin_amdgcn_wmma_f32_16x16x32_bf16(false, afA.h, false, b1.h, (short)0, a1, false, false);
        a2 = __builtin_amdgcn_wmma_f32_16x16x32_bf16(false, afA.h, false, b2.h, (short)0, a2, false, false);
        a3 = __builtin_amdgcn_wmma_f32_16x16x32_bf16(false, afA.h, false, b3.h, (short)0, a3, false, false);
        c0 = __builtin_amdgcn_wmma_f32_16x16x32_bf16(false, afB.h, false, b0.h, (short)0, c0, false, false);
        c1 = __builtin_amdgcn_wmma_f32_16x16x32_bf16(false, afB.h, false, b1.h, (short)0, c1, false, false);
        c2 = __builtin_amdgcn_wmma_f32_16x16x32_bf16(false, afB.h, false, b2.h, (short)0, c2, false, false);
        c3 = __builtin_amdgcn_wmma_f32_16x16x32_bf16(false, afB.h, false, b3.h, (short)0, c3, false, false);
    }

    // C/D layout: VGPR i -> row offset i + 8*half, col = tn*64 + nt*16 + r
    const int cb = tn * 64 + r;
    const float bv0 = bias ? bias[cb]      : 0.f;
    const float bv1 = bias ? bias[cb + 16] : 0.f;
    const float bv2 = bias ? bias[cb + 32] : 0.f;
    const float bv3 = bias ? bias[cb + 48] : 0.f;
#pragma unroll
    for (int i = 0; i < 8; ++i) {
        int row = tm * 32 + i + half * 8;
        if (row < M) {
            float* crow = C + (size_t)row * N;
            if (accumulate) {
                crow[cb]      += a0[i] + bv0; crow[cb + 16] += a1[i] + bv1;
                crow[cb + 32] += a2[i] + bv2; crow[cb + 48] += a3[i] + bv3;
            } else {
                crow[cb]      = a0[i] + bv0;  crow[cb + 16] = a1[i] + bv1;
                crow[cb + 32] = a2[i] + bv2;  crow[cb + 48] = a3[i] + bv3;
            }
        }
        int row2 = row + 16;
        if (row2 < M) {
            float* crow = C + (size_t)row2 * N;
            if (accumulate) {
                crow[cb]      += c0[i] + bv0; crow[cb + 16] += c1[i] + bv1;
                crow[cb + 32] += c2[i] + bv2; crow[cb + 48] += c3[i] + bv3;
            } else {
                crow[cb]      = c0[i] + bv0;  crow[cb + 16] = c1[i] + bv1;
                crow[cb + 32] = c2[i] + bv2;  crow[cb + 48] = c3[i] + bv3;
            }
        }
    }
}

// ---------------------------------------------------------------------------
// Small helper kernels
// ---------------------------------------------------------------------------
__global__ void cvt_f32_bf16(const float* __restrict__ src, unsigned short* __restrict__ dst, int n) {
    int t = blockIdx.x * 256 + threadIdx.x;
    if (t < n) dst[t] = f2bf(src[t]);
}

__global__ void in_proj_kernel(const float* __restrict__ X, const float* __restrict__ W,
                               const float* __restrict__ b,
                               float* __restrict__ Y32, unsigned short* __restrict__ Yb,
                               int Nn, int Kin)
{
    int t = blockIdx.x * 256 + threadIdx.x;
    if (t >= Nn * 64) return;
    int n = t >> 6, c = t & 63;
    const float* x = X + (size_t)n * Kin;
    float a = b[c];
    for (int i = 0; i < Kin; ++i) a = fmaf(x[i], W[i * 64 + c], a);
    a = fmaxf(a, 0.f);
    Y32[t] = a;
    Yb[t]  = f2bf(a);
}

__global__ void edge_proj_kernel(const float* __restrict__ EA, const float* __restrict__ W,
                                 const float* __restrict__ b, float* __restrict__ EP,
                                 int Ecnt, int Kin)
{
    int e = blockIdx.x * 256 + threadIdx.x;
    if (e >= Ecnt) return;
#pragma unroll
    for (int d = 0; d < 3; ++d) {
        float a = b[d];
        for (int i = 0; i < Kin; ++i) a = fmaf(EA[(size_t)e * Kin + i], W[i * 3 + d], a);
        EP[(size_t)e * 3 + d] = a;
    }
}

__global__ void init_ms(float* __restrict__ m, float* __restrict__ s, int n) {
    int t = blockIdx.x * 256 + threadIdx.x;
    if (t >= n) return;
    m[t] = __int_as_float(0xff800000);   // -inf
    s[t] = 0.f;
}

// pass 1: alpha = <q[dst], k[src]+e> / sqrt(64); per-(dst,head) atomic max
__global__ void edge_alpha(const int* __restrict__ src, const int* __restrict__ dst,
                           const float* __restrict__ ep, const float* __restrict__ We,
                           const float* __restrict__ Q, const float* __restrict__ Kf,
                           float* __restrict__ alpha, float* __restrict__ mbuf, int Ecnt)
{
    int t = blockIdx.x * 256 + threadIdx.x;
    if (t >= Ecnt * 4) return;
    int e = t >> 2, h = t & 3;
    int si = src[e], di = dst[e];
    float e0 = ep[(size_t)e * 3], e1 = ep[(size_t)e * 3 + 1], e2 = ep[(size_t)e * 3 + 2];
    const v4f* q4 = (const v4f*)(Q  + (size_t)di * 256 + h * 64);
    const v4f* k4 = (const v4f*)(Kf + (size_t)si * 256 + h * 64);
    const v4f* w4 = (const v4f*)(We + h * 64);
    float a = 0.f;
#pragma unroll 4
    for (int c = 0; c < 16; ++c) {
        v4f kv = k4[c], qv = q4[c];
        v4f w0 = w4[c], w1 = w4[64 + c], w2 = w4[128 + c];
#pragma unroll
        for (int i = 0; i < 4; ++i) {
            float kj = kv[i] + e0 * w0[i] + e1 * w1[i] + e2 * w2[i];
            a = fmaf(qv[i], kj, a);
        }
    }
    a *= 0.125f;                          // 1/sqrt(64)
    alpha[t] = a;
    atomicMaxF(&mbuf[(size_t)di * 4 + h], a);
}

// pass 2: ex = exp(alpha - m); per-(dst,head) atomic sum
__global__ void edge_soft(const int* __restrict__ dst, const float* __restrict__ m,
                          float* __restrict__ s, float* __restrict__ alpha, int Ecnt)
{
    int t = blockIdx.x * 256 + threadIdx.x;
    if (t >= Ecnt * 4) return;
    int e = t >> 2, h = t & 3;
    int di = dst[e];
    float ex = __expf(alpha[t] - m[(size_t)di * 4 + h]);
    alpha[t] = ex;
    atomicAdd(&s[(size_t)di * 4 + h], ex);
}

// pass 3: OUT[dst] += (v[src]+e) * ex/(sum+eps)
__global__ void edge_scatter(const int* __restrict__ src, const int* __restrict__ dst,
                             const float* __restrict__ ep, const float* __restrict__ We,
                             const float* __restrict__ V, const float* __restrict__ alpha,
                             const float* __restrict__ s, float* __restrict__ OUT, int Ecnt)
{
    int t = blockIdx.x * 256 + threadIdx.x;
    if (t >= Ecnt * 4) return;
    int e = t >> 2, h = t & 3;
    int si = src[e], di = dst[e];
    float coef = alpha[t] / (s[(size_t)di * 4 + h] + 1e-16f);
    float e0 = ep[(size_t)e * 3], e1 = ep[(size_t)e * 3 + 1], e2 = ep[(size_t)e * 3 + 2];
    const v4f* v4 = (const v4f*)(V + (size_t)si * 256 + h * 64);
    const v4f* w4 = (const v4f*)(We + h * 64);
    float* o = OUT + (size_t)di * 256 + h * 64;
#pragma unroll 4
    for (int c = 0; c < 16; ++c) {
        v4f vv = v4[c];
        v4f w0 = w4[c], w1 = w4[64 + c], w2 = w4[128 + c];
#pragma unroll
        for (int i = 0; i < 4; ++i) {
            float ef = e0 * w0[i] + e1 * w1[i] + e2 * w2[i];
            atomicAdd(&o[c * 4 + i], (vv[i] + ef) * coef);
        }
    }
}

// elu + residual + LayerNorm(64) fused; one wave32 per node, 2 features/lane
__global__ __launch_bounds__(256)
void elu_res_ln(const float* __restrict__ Hp, const float* __restrict__ g,
                const float* __restrict__ bb,
                float* __restrict__ X32, unsigned short* __restrict__ Xb, int Nn)
{
    int wave = threadIdx.x >> 5, lane = threadIdx.x & 31;
    int n = blockIdx.x * 8 + wave;
    if (n >= Nn) return;
    const float* hp = Hp  + (size_t)n * 64;
    float*       xp = X32 + (size_t)n * 64;

    float h0 = hp[lane], h1 = hp[lane + 32];
    h0 = h0 > 0.f ? h0 : __expf(h0) - 1.f;
    h1 = h1 > 0.f ? h1 : __expf(h1) - 1.f;
    float v0 = xp[lane] + h0, v1 = xp[lane + 32] + h1;

    float ssum = v0 + v1;
#pragma unroll
    for (int off = 16; off; off >>= 1) ssum += __shfl_xor(ssum, off, 32);
    float mu = ssum * (1.f / 64.f);
    float d0 = v0 - mu, d1 = v1 - mu;
    float vs = d0 * d0 + d1 * d1;
#pragma unroll
    for (int off = 16; off; off >>= 1) vs += __shfl_xor(vs, off, 32);
    float inv = rsqrtf(vs * (1.f / 64.f) + 1e-5f);

    float y0 = d0 * inv * g[lane]      + bb[lane];
    float y1 = d1 * inv * g[lane + 32] + bb[lane + 32];
    xp[lane] = y0;  xp[lane + 32] = y1;
    Xb[(size_t)n * 64 + lane]      = f2bf(y0);
    Xb[(size_t)n * 64 + lane + 32] = f2bf(y1);
}

// head tail: out[n] = b2 + sum_c relu(H1[n,c]) * W2[c]
__global__ void head_out(const float* __restrict__ H1, const float* __restrict__ W2,
                         const float* __restrict__ b2, float* __restrict__ out, int Nn)
{
    int n = blockIdx.x * 256 + threadIdx.x;
    if (n >= Nn) return;
    const v4f* h4 = (const v4f*)(H1 + (size_t)n * 64);
    const v4f* w4 = (const v4f*)W2;
    float a = b2[0];
#pragma unroll
    for (int c = 0; c < 16; ++c) {
        v4f hv = h4[c], wv = w4[c];
#pragma unroll
        for (int i = 0; i < 4; ++i) a = fmaf(fmaxf(hv[i], 0.f), wv[i], a);
    }
    out[n] = a;
}

// ---------------------------------------------------------------------------
extern "C" void kernel_launch(void* const* d_in, const int* in_sizes, int n_in,
                              void* d_out, int out_size, void* d_ws, size_t ws_size,
                              hipStream_t stream)
{
    (void)n_in; (void)out_size; (void)ws_size;
    const int NG = in_sizes[0] / 16;
    const int NS = in_sizes[1] / 8;
    const int E  = in_sizes[5];

    const float* x_grid   = (const float*)d_in[0];
    const float* x_sensor = (const float*)d_in[1];
    const float* ea_gg    = (const float*)d_in[2];
    const float* ea_gs    = (const float*)d_in[3];
    const float* ea_sg    = (const float*)d_in[4];
    const int*   src_gg   = (const int*)d_in[5];
    const int*   dst_gg   = (const int*)d_in[6];
    const int*   src_gs   = (const int*)d_in[7];
    const int*   dst_gs   = (const int*)d_in[8];
    const int*   src_sg   = (const int*)d_in[9];
    const int*   dst_sg   = (const int*)d_in[10];

    // params flattened in sorted-key pytree order starting at index 11
    const int P0 = 11;
    const int EPI = P0;        // edge_proj: gg.W, gg.b, gs.W, gs.b, sg.W, sg.b
    const int HD  = P0 + 6;    // head: grid.l1.W,l1.b,l2.W,l2.b, sensor.l1.W,l1.b,l2.W,l2.b
    const int INP = P0 + 14;   // in_proj: grid.W, grid.b, sensor.W, sensor.b
    const int LYR = P0 + 18;   // layers, stride 35
    auto F = [&](int i) -> const float* { return (const float*)d_in[i]; };

    // -------- workspace carve --------
    char* wp = (char*)d_ws;
    auto alloc = [&](size_t bytes) -> void* {
        void* r = (void*)wp; wp += (bytes + 255) & ~(size_t)255; return r;
    };
    float*          xg32  = (float*)alloc((size_t)NG * 64 * 4);
    float*          xs32  = (float*)alloc((size_t)NS * 64 * 4);
    unsigned short* xgb   = (unsigned short*)alloc((size_t)NG * 64 * 2);
    unsigned short* xsb   = (unsigned short*)alloc((size_t)NS * 64 * 2);
    float*          Qb    = (float*)alloc((size_t)NG * 256 * 4);
    float*          Kb    = (float*)alloc((size_t)NG * 256 * 4);
    float*          Vb    = (float*)alloc((size_t)NG * 256 * 4);
    float*          OUTg  = (float*)alloc((size_t)NG * 256 * 4);
    float*          OUTs  = (float*)alloc((size_t)NS * 256 * 4);
    float*          alphab= (float*)alloc((size_t)E * 4 * 4);
    float*          mbuf  = (float*)alloc((size_t)NG * 4 * 4);
    float*          sbuf  = (float*)alloc((size_t)NG * 4 * 4);
    float*          epgg  = (float*)alloc((size_t)E * 3 * 4);
    float*          epgs  = (float*)alloc((size_t)E * 3 * 4);
    float*          epsg  = (float*)alloc((size_t)E * 3 * 4);
    float*          Hproj = (float*)alloc((size_t)NG * 64 * 4);
    unsigned short* wpool = (unsigned short*)alloc((size_t)(24 * 16384 + 4 * 16384 + 2 * 4096) * 2);
    // conv outputs in bf16 reuse dead K/V buffers
    unsigned short* OUTg_bf = (unsigned short*)Vb;
    unsigned short* OUTs_bf = (unsigned short*)Kb;

    size_t woff = 0;
    auto packW = [&](const float* src, int K, int N) -> unsigned short* {
        unsigned short* dst = wpool + woff; woff += (size_t)K * N;
        pack_b_wmma<<<(K * N + 255) / 256, 256, 0, stream>>>(src, dst, K, N);
        return dst;
    };
    auto gemm = [&](const unsigned short* A, const unsigned short* B, const float* bias,
                    float* C, int M, int N, int K, int acc) {
        int tiles  = ((M + 31) / 32) * (N / 64);
        int blocks = (tiles + 3) / 4;
        gemm_bf16_wmma<<<blocks, 128, 0, stream>>>(A, B, bias, C, M, N, K, acc);
    };

    // -------- bf16 weight convert+pack (deterministic, every launch) --------
    struct ConvW { unsigned short *kW, *qW, *sW, *vW; };
    ConvW cw[2][3];
    unsigned short* projW[2][2];
    for (int l = 0; l < 2; ++l) {
        int base = LYR + 35 * l;
        for (int c = 0; c < 3; ++c) {
            int cb = base + 9 * c;
            cw[l][c].kW = packW(F(cb + 1), 64, 256);
            cw[l][c].qW = packW(F(cb + 3), 64, 256);
            cw[l][c].sW = packW(F(cb + 5), 64, 256);
            cw[l][c].vW = packW(F(cb + 7), 64, 256);
        }
        projW[l][0] = packW(F(base + 31), 256, 64);
        projW[l][1] = packW(F(base + 33), 256, 64);
    }
    unsigned short* headW[2] = { packW(F(HD + 0), 64, 64), packW(F(HD + 4), 64, 64) };

    // -------- input / edge projections --------
    in_proj_kernel<<<((NG * 64) + 255) / 256, 256, 0, stream>>>(x_grid,   F(INP + 0), F(INP + 1), xg32, xgb, NG, 16);
    in_proj_kernel<<<((NS * 64) + 255) / 256, 256, 0, stream>>>(x_sensor, F(INP + 2), F(INP + 3), xs32, xsb, NS, 8);
    edge_proj_kernel<<<(E + 255) / 256, 256, 0, stream>>>(ea_gg, F(EPI + 0), F(EPI + 1), epgg, E, 3);
    edge_proj_kernel<<<(E + 255) / 256, 256, 0, stream>>>(ea_gs, F(EPI + 2), F(EPI + 3), epgs, E, 1);
    edge_proj_kernel<<<(E + 255) / 256, 256, 0, stream>>>(ea_sg, F(EPI + 4), F(EPI + 5), epsg, E, 1);

    const int eb = (E * 4 + 255) / 256;
    auto run_conv = [&](const unsigned short* xsrc, const unsigned short* xdst,
                        int n_src, int n_dst,
                        const int* srcI, const int* dstI, const float* ep,
                        const float* We, const ConvW& w,
                        const float* kbias, const float* qbias,
                        const float* sbias, const float* vbias,
                        float* OUT, int acc) {
        gemm(xdst, w.qW, qbias, Qb, n_dst, 256, 64, 0);
        gemm(xsrc, w.kW, kbias, Kb, n_src, 256, 64, 0);
        gemm(xsrc, w.vW, vbias, Vb, n_src, 256, 64, 0);
        gemm(xdst, w.sW, sbias, OUT, n_dst, 256, 64, acc);
        init_ms<<<(n_dst * 4 + 255) / 256, 256, 0, stream>>>(mbuf, sbuf, n_dst * 4);
        edge_alpha  <<<eb, 256, 0, stream>>>(srcI, dstI, ep, We, Qb, Kb, alphab, mbuf, E);
        edge_soft   <<<eb, 256, 0, stream>>>(dstI, mbuf, sbuf, alphab, E);
        edge_scatter<<<eb, 256, 0, stream>>>(srcI, dstI, ep, We, Vb, alphab, sbuf, OUT, E);
    };

    // -------- layers --------
    for (int l = 0; l < 2; ++l) {
        int base = LYR + 35 * l;
        int gg = base + 0, gs = base + 9, sg = base + 18;
        // conv gg: grid -> grid (OUTg = store)
        run_conv(xgb, xgb, NG, NG, src_gg, dst_gg, epgg, F(gg + 0), cw[l][0],
                 F(gg + 2), F(gg + 4), F(gg + 6), F(gg + 8), OUTg, 0);
        // conv sg: sensor -> grid (OUTg += )
        run_conv(xsb, xgb, NS, NG, src_sg, dst_sg, epsg, F(sg + 0), cw[l][2],
                 F(sg + 2), F(sg + 4), F(sg + 6), F(sg + 8), OUTg, 1);
        // conv gs: grid -> sensor (OUTs = store)
        run_conv(xgb, xsb, NG, NS, src_gs, dst_gs, epgs, F(gs + 0), cw[l][1],
                 F(gs + 2), F(gs + 4), F(gs + 6), F(gs + 8), OUTs, 0);

        // conv outputs -> bf16 (K/V buffers are dead now)
        cvt_f32_bf16<<<((NG * 256) + 255) / 256, 256, 0, stream>>>(OUTg, OUTg_bf, NG * 256);
        cvt_f32_bf16<<<((NS * 256) + 255) / 256, 256, 0, stream>>>(OUTs, OUTs_bf, NS * 256);

        // proj + elu + residual + LN (grid, then sensor)
        gemm(OUTg_bf, projW[l][0], F(base + 32), Hproj, NG, 64, 256, 0);
        elu_res_ln<<<(NG + 7) / 8, 256, 0, stream>>>(Hproj, F(base + 28), F(base + 27), xg32, xgb, NG);
        gemm(OUTs_bf, projW[l][1], F(base + 34), Hproj, NS, 64, 256, 0);
        elu_res_ln<<<(NS + 7) / 8, 256, 0, stream>>>(Hproj, F(base + 30), F(base + 29), xs32, xsb, NS);
    }

    // -------- heads --------
    float* out_g = (float*)d_out;
    float* out_s = out_g + NG;
    gemm(xgb, headW[0], F(HD + 1), Hproj, NG, 64, 64, 0);
    head_out<<<(NG + 255) / 256, 256, 0, stream>>>(Hproj, F(HD + 2), F(HD + 3), out_g, NG);
    gemm(xsb, headW[1], F(HD + 5), Hproj, NS, 64, 64, 0);
    head_out<<<(NS + 255) / 256, 256, 0, stream>>>(Hproj, F(HD + 6), F(HD + 7), out_s, NS);
}
